// GCNRegression_79139067396492
// MI455X (gfx1250) — matrix operand order repaired and checked by reading, not compiled, not run on hardware
//
#include <hip/hip_runtime.h>

typedef __attribute__((ext_vector_type(16))) __bf16 v16bf;
typedef __attribute__((ext_vector_type(8)))  __bf16 v8bf;
typedef __attribute__((ext_vector_type(8)))  float  v8f;

#define FDIM 128

// round-to-nearest-even fp32 -> bf16 (bit-exact, no reliance on cast codegen)
__device__ __forceinline__ __bf16 f2bf(float f) {
    union { float f; unsigned u; } in; in.f = f;
    unsigned u = in.u;
    unsigned r = u + 0x7FFFu + ((u >> 16) & 1u);
    unsigned short h = (unsigned short)(r >> 16);
    __bf16 out;
    __builtin_memcpy(&out, &h, sizeof(out));
    return out;
}

// ---------------- utility: zero a float buffer (float4 vectorized) ----------
__global__ __launch_bounds__(256) void zero_f32(float* __restrict__ p, int n) {
    int i4 = (blockIdx.x * 256 + threadIdx.x) * 4;
    if (i4 >= n) return;
    if (i4 + 4 <= n) {
        *(float4*)(p + i4) = make_float4(0.f, 0.f, 0.f, 0.f);
    } else {
        for (int j = i4; j < n; ++j) p[j] = 0.f;
    }
}

// ---------------- degree counting (float atomics, exact for int counts) ----
__global__ __launch_bounds__(256) void degree_kernel(
    const int* __restrict__ src, const int* __restrict__ dst,
    float* __restrict__ out_deg, float* __restrict__ in_deg, int nE) {
    int e = blockIdx.x * 256 + threadIdx.x;
    if (e >= nE) return;
    atomicAdd(&out_deg[src[e]], 1.0f);
    atomicAdd(&in_deg[dst[e]], 1.0f);
}

// deg -> clip(deg,1)^-0.5 in place
__global__ __launch_bounds__(256) void to_norm(float* __restrict__ p, int n) {
    int i = blockIdx.x * 256 + threadIdx.x;
    if (i >= n) return;
    float d = p[i];
    d = d < 1.0f ? 1.0f : d;
    p[i] = rsqrtf(d);
}

// ---------------- fused (prev-layer finish) + GEMM + source-norm ------------
// x[n,:] = ( act( hin[n,:] * pre_norm[n] + bias_prev ) @ W ) * out_norm[n]
// A tile (16 rows) staged to LDS as bf16 with the previous layer's epilogue
// fused in; W staged transposed as bf16. 8 waves/block, one 16-col N-tile
// each, K=128 via 4x v_wmma_f32_16x16x32_bf16 with fp32 accumulation.
// Result goes back through LDS so global stores are coalesced float4s.
__global__ __launch_bounds__(256) void gemm_pre_bf16(
    const float* __restrict__ hin,        // [N,128] features or prev agg
    const float* __restrict__ pre_norm,   // in_norm (used when apply_pre)
    const float* __restrict__ bias_prev,  // [128]   (used when apply_pre)
    int apply_pre,
    const float* __restrict__ Wmat,       // [128,128] row-major (k, c)
    const float* __restrict__ out_norm,   // [N]
    float* __restrict__ xout,             // [N,128]
    int nN) {
    __shared__ __bf16 sA[16][FDIM];       // A tile, row-major
    __shared__ __bf16 sWT[FDIM][FDIM];    // W transposed: sWT[c][k]
    __shared__ float  sOut[16][FDIM];     // f32 result staging
    __shared__ float  sOn[16];

    const int tid   = threadIdx.x;
    const int mbase = blockIdx.x * 16;

    // stage W (transposed) as bf16: 64 elems/thread, L2-resident reads
    for (int i = tid; i < FDIM * FDIM; i += 256) {
        int c = i >> 7, k = i & 127;
        sWT[c][k] = f2bf(Wmat[k * FDIM + c]);
    }
    // stage A tile with fused norm+bias+relu of the previous layer
    for (int i = tid; i < 16 * FDIM; i += 256) {
        int r = i >> 7, k = i & 127;
        int node = mbase + r;
        float v = 0.f;
        if (node < nN) {
            v = hin[(size_t)node * FDIM + k];
            if (apply_pre) {
                v = v * pre_norm[node] + bias_prev[k];
                v = v > 0.f ? v : 0.f;          // layers 1..5 all have ReLU
            }
        }
        sA[r][k] = f2bf(v);
    }
    if (tid < 16) {
        int node = mbase + tid;
        sOn[tid] = (node < nN) ? out_norm[node] : 0.f;
    }
    __syncthreads();

    const int wave = tid >> 5;           // 0..7 -> N-tile
    const int lane = tid & 31;
    const int half = lane >> 4;          // 0: lanes 0-15, 1: lanes 16-31
    const int row  = lane & 15;          // A row within tile / D column
    const int akb  = half ? 16 : 0;      // A K-half select (contiguous, matches B)
    const int bkb  = half ? 16 : 0;      // B K-half select (contiguous, matches A)

    v8f acc = {};
#pragma unroll
    for (int k0 = 0; k0 < FDIM; k0 += 32) {
        // A lane data: K = k0+akb+{0..7} in v[0..7], k0+akb+8+{0..7} in v[8..15]
        v8bf alo = *(const v8bf*)&sA[row][k0 + akb];
        v8bf ahi = *(const v8bf*)&sA[row][k0 + akb + 8];
        v16bf a  = __builtin_shufflevector(alo, ahi,
                                           0, 1, 2, 3, 4, 5, 6, 7,
                                           8, 9, 10, 11, 12, 13, 14, 15);
        // B lane data: column = lane&15, K = k0+bkb .. k0+bkb+15 (contiguous)
        v16bf b = *(const v16bf*)&sWT[wave * 16 + row][k0 + bkb];
        acc = __builtin_amdgcn_wmma_f32_16x16x32_bf16(
            false, a, false, b, (short)0, acc, false, false);
    }

    // D layout: VGPR r holds M = r + 8*half, N = lane&15 -> stage to LDS
#pragma unroll
    for (int r = 0; r < 8; ++r) {
        int m = r + half * 8;
        sOut[m][wave * 16 + row] = acc[r];
    }
    __syncthreads();

    // coalesced writeback: 16x128 floats, float4 per thread (2 iterations)
    for (int i = tid; i < 16 * (FDIM / 4); i += 256) {
        int r  = i >> 5;                 // row 0..15
        int c4 = (i & 31) * 4;           // column group
        int node = mbase + r;
        if (node < nN) {
            float on = sOn[r];
            float4 v;
            v.x = sOut[r][c4 + 0] * on;
            v.y = sOut[r][c4 + 1] * on;
            v.z = sOut[r][c4 + 2] * on;
            v.w = sOut[r][c4 + 3] * on;
            *(float4*)&xout[(size_t)node * FDIM + c4] = v;
        }
    }
}

// ---------------- SpMM scatter: agg[dst] += x[src] --------------------------
// One wave32 per edge: indices forced to SGPRs (scalar loads + saddr-form
// vector ops), float4 gather of the 512B row (L2-resident), then 4
// global_atomic_add_f32 per lane.
__global__ __launch_bounds__(256) void spmm_scatter(
    const float* __restrict__ x, const int* __restrict__ src,
    const int* __restrict__ dst, float* __restrict__ agg, int nE) {
    int t = blockIdx.x * 256 + threadIdx.x;
    int e = t >> 5;
    if (e >= nE) return;
    e = __builtin_amdgcn_readfirstlane(e);   // wave-uniform: one edge per wave
    int lane = t & 31;
    int s = __builtin_amdgcn_readfirstlane(src[e]);
    int d = __builtin_amdgcn_readfirstlane(dst[e]);
    float4 v = *(const float4*)(x + (size_t)s * FDIM + lane * 4);
    float* a = agg + (size_t)d * FDIM + lane * 4;
    atomicAdd(a + 0, v.x);
    atomicAdd(a + 1, v.y);
    atomicAdd(a + 2, v.z);
    atomicAdd(a + 3, v.w);
}

// ---------------- final: out[n] = (finish6(agg)[n,:] @ Wl) + bl -------------
// one wave32 per node; layer 6 has no ReLU.
__global__ __launch_bounds__(256) void final_linear(
    const float* __restrict__ agg, const float* __restrict__ in_norm,
    const float* __restrict__ b6, const float* __restrict__ Wl,
    const float* __restrict__ bl, float* __restrict__ out, int nN) {
    int wid  = (blockIdx.x * 256 + threadIdx.x) >> 5;
    int lane = threadIdx.x & 31;
    if (wid >= nN) return;
    float inn = in_norm[wid];
    float acc = 0.f;
#pragma unroll
    for (int j = 0; j < 4; ++j) {
        int f = j * 32 + lane;
        float v = agg[(size_t)wid * FDIM + f] * inn + b6[f];
        acc += v * Wl[f];
    }
#pragma unroll
    for (int off = 16; off > 0; off >>= 1)
        acc += __shfl_down(acc, off, 32);
    if (lane == 0) out[wid] = acc + bl[0];
}

extern "C" void kernel_launch(void* const* d_in, const int* in_sizes, int n_in,
                              void* d_out, int out_size, void* d_ws, size_t ws_size,
                              hipStream_t stream) {
    const float* features = (const float*)d_in[0];
    const int*   src      = (const int*)d_in[1];
    const int*   dst      = (const int*)d_in[2];
    const float* Wt[6] = {(const float*)d_in[3],  (const float*)d_in[5],
                          (const float*)d_in[7],  (const float*)d_in[9],
                          (const float*)d_in[11], (const float*)d_in[13]};
    const float* bt[6] = {(const float*)d_in[4],  (const float*)d_in[6],
                          (const float*)d_in[8],  (const float*)d_in[10],
                          (const float*)d_in[12], (const float*)d_in[14]};
    const float* Wl = (const float*)d_in[15];
    const float* bl = (const float*)d_in[16];
    float* out = (float*)d_out;

    const int nN = in_sizes[0] / FDIM;   // 100000
    const int nE = in_sizes[1];          // 1600000

    // workspace carve-up (aligned to 256B)
    char* ws = (char*)d_ws;
    auto carve = [&](size_t bytes) {
        char* p = ws;
        ws += (bytes + 255) & ~(size_t)255;
        return p;
    };
    float* norms    = (float*)carve((size_t)2 * nN * sizeof(float)); // [out|in]
    float* out_norm = norms;
    float* in_norm  = norms + nN;
    float* xbuf     = (float*)carve((size_t)nN * FDIM * sizeof(float));
    float* aggbuf   = (float*)carve((size_t)nN * FDIM * sizeof(float));

    const int T = 256;
    int gZn  = (2 * nN + (T * 4) - 1) / (T * 4);              // zero norms
    int gDeg = (nE + T - 1) / T;
    int gNrm = (2 * nN + T - 1) / T;
    int gGemm = (nN + 15) / 16;
    int gZa  = (nN * FDIM + (T * 4) - 1) / (T * 4);           // zero agg
    int gSp  = (int)(((long long)nE * 32 + T - 1) / T);
    int gFin = (nN * 32 + T - 1) / T;

    // degrees -> norms
    zero_f32<<<gZn, T, 0, stream>>>(norms, 2 * nN);
    degree_kernel<<<gDeg, T, 0, stream>>>(src, dst, out_norm, in_norm, nE);
    to_norm<<<gNrm, T, 0, stream>>>(norms, 2 * nN);

    // 6 GraphConv layers; previous layer's finish (in_norm, bias, relu) is
    // fused into the next GEMM's A-tile staging.
    const float* hsrc = features;
    for (int L = 0; L < 6; ++L) {
        gemm_pre_bf16<<<gGemm, T, 0, stream>>>(
            hsrc, in_norm, L ? bt[L - 1] : bt[0], L > 0 ? 1 : 0,
            Wt[L], out_norm, xbuf, nN);
        zero_f32<<<gZa, T, 0, stream>>>(aggbuf, nN * FDIM);
        spmm_scatter<<<gSp, T, 0, stream>>>(xbuf, src, dst, aggbuf, nE);
        hsrc = aggbuf;
    }

    // layer-6 finish (no relu) + 128->1 linear
    final_linear<<<gFin, T, 0, stream>>>(aggbuf, in_norm, bt[5], Wl, bl, out, nN);
}
